// MambaBlock_43843026158205
// MI455X (gfx1250) — matrix operand order repaired and checked by reading, not compiled, not run on hardware
//
#include <hip/hip_runtime.h>
#include <hip/hip_bf16.h>

// ---------------------------------------------------------------------------
// Mamba block for gfx1250 (MI455X). Big GEMMs via v_wmma_f32_16x16x32_bf16.
// B=2, L=2048, DM=1024, DS=16, DC=4, EXP=2, DI=2048.
// ---------------------------------------------------------------------------

typedef __bf16 bf16;
typedef __attribute__((ext_vector_type(8)))  __bf16 v8bf;
typedef __attribute__((ext_vector_type(16))) __bf16 v16bf;
typedef __attribute__((ext_vector_type(8)))  float  v8f;

#define B_   2
#define L_   2048
#define DM_  1024
#define DS_  16
#define DC_  4
#define DI_  2048
#define NC_  4096          // 2*DI
#define ROWS_ 4096         // B*L
#define NPROJ_ 33          // 2*DS + 1

// ---------------------------------------------------------------------------
// fp32 -> bf16 conversion
// ---------------------------------------------------------------------------
__global__ void f32_to_bf16_kernel(const float* __restrict__ in,
                                   bf16* __restrict__ out, int n) {
    int i = blockIdx.x * blockDim.x + threadIdx.x;
    if (i < n) out[i] = (bf16)in[i];
}

__device__ __forceinline__ v16bf cat8(v8bf lo, v8bf hi) {
    return __builtin_shufflevector(lo, hi, 0, 1, 2, 3, 4, 5, 6, 7,
                                           8, 9, 10, 11, 12, 13, 14, 15);
}

// ---------------------------------------------------------------------------
// NT GEMM: C[M,N] = A[M,K] * W[N,K]^T, bf16 in, fp32 accumulate.
// Fragment layouts as loaded here (wave32, 16-bit elements; contiguous k, not the ISA table order):
//   A 16x32 (MxK): lane l (l<16) holds row l, K = [0..15] contiguous;
//                  lane l+16 holds row l, K = [16..31] contiguous.
//   B 32x16 (KxN): lane l holds column (l&15), K = (l>>4)*16 .. +15 contiguous.
// Block = 128 threads (4 waves, 2x2): block tile 64x128, wave tile 32x64
// (2x4 WMMA tiles -> 8 independent v_wmma per k-step). Six explicit pointer
// induction variables keep addressing out of scratch; launch_bounds(128,1)
// gives the allocator the full VGPR budget (no spills).
// Grid: (N/128, M/64).
// ---------------------------------------------------------------------------
template <bool OUT_BF16>
__global__ void __launch_bounds__(128, 1)
wmma_gemm_nt(const bf16* __restrict__ A, const bf16* __restrict__ W,
             float* __restrict__ Cf, bf16* __restrict__ Cb,
             int M, int N, int K) {
    const int lane  = threadIdx.x & 31;
    const int wave  = threadIdx.x >> 5;
    const int mBase = blockIdx.y * 64  + (wave >> 1) * 32;
    const int nBase = blockIdx.x * 128 + (wave & 1) * 64;

    // Per-lane stream pointers (lo chunk; hi chunk via immediate offset).
    const int r16 = lane & 15;
    const bf16* pA0 = A + (size_t)(mBase + r16) * K + (lane >> 4) * 16;
    const bf16* pA1 = pA0 + (size_t)16 * K;
    const bf16* pB0 = W + (size_t)(nBase + r16) * K + (lane >> 4) * 16;
    const bf16* pB1 = pB0 + (size_t)16 * K;
    const bf16* pB2 = pB1 + (size_t)16 * K;
    const bf16* pB3 = pB2 + (size_t)16 * K;

    v8f acc[2][4] = {};

    for (int k0 = 0; k0 < K; k0 += 32) {
        // Speculative L2 hint for the next K tile (global_prefetch_b8);
        // invalid translations are silently dropped, so no guard branch.
        __builtin_prefetch(pA0 + 32, 0, 1);
        __builtin_prefetch(pB0 + 32, 0, 1);

        v16bf a0 = cat8(*(const v8bf*)pA0, *(const v8bf*)(pA0 + 8));
        v16bf a1 = cat8(*(const v8bf*)pA1, *(const v8bf*)(pA1 + 8));
        v16bf b0 = cat8(*(const v8bf*)pB0, *(const v8bf*)(pB0 + 8));
        v16bf b1 = cat8(*(const v8bf*)pB1, *(const v8bf*)(pB1 + 8));
        v16bf b2 = cat8(*(const v8bf*)pB2, *(const v8bf*)(pB2 + 8));
        v16bf b3 = cat8(*(const v8bf*)pB3, *(const v8bf*)(pB3 + 8));

        acc[0][0] = __builtin_amdgcn_wmma_f32_16x16x32_bf16(false, a0, false, b0,
                                                            (short)0, acc[0][0], false, false);
        acc[0][1] = __builtin_amdgcn_wmma_f32_16x16x32_bf16(false, a0, false, b1,
                                                            (short)0, acc[0][1], false, false);
        acc[0][2] = __builtin_amdgcn_wmma_f32_16x16x32_bf16(false, a0, false, b2,
                                                            (short)0, acc[0][2], false, false);
        acc[0][3] = __builtin_amdgcn_wmma_f32_16x16x32_bf16(false, a0, false, b3,
                                                            (short)0, acc[0][3], false, false);
        acc[1][0] = __builtin_amdgcn_wmma_f32_16x16x32_bf16(false, a1, false, b0,
                                                            (short)0, acc[1][0], false, false);
        acc[1][1] = __builtin_amdgcn_wmma_f32_16x16x32_bf16(false, a1, false, b1,
                                                            (short)0, acc[1][1], false, false);
        acc[1][2] = __builtin_amdgcn_wmma_f32_16x16x32_bf16(false, a1, false, b2,
                                                            (short)0, acc[1][2], false, false);
        acc[1][3] = __builtin_amdgcn_wmma_f32_16x16x32_bf16(false, a1, false, b3,
                                                            (short)0, acc[1][3], false, false);

        pA0 += 32; pA1 += 32;
        pB0 += 32; pB1 += 32; pB2 += 32; pB3 += 32;
    }

    // C/D layout: VGPR r, lanes 0-15 -> row r, lanes 16-31 -> row r+8; col = lane&15
    const int col    = lane & 15;
    const int rowOff = (lane >> 4) * 8;
#pragma unroll
    for (int tm = 0; tm < 2; ++tm) {
#pragma unroll
        for (int tn = 0; tn < 4; ++tn) {
            const v8f a = acc[tm][tn];
#pragma unroll
            for (int r = 0; r < 8; ++r) {
                size_t m = (size_t)(mBase + tm * 16 + rowOff + r);
                size_t n = (size_t)(nBase + tn * 16 + col);
                if (OUT_BF16) Cb[m * N + n] = (bf16)a[r];
                else          Cf[m * N + n] = a[r];
            }
        }
    }
}

// ---------------------------------------------------------------------------
// Depthwise causal conv (DC=4) + bias, gated by silu(z). XZ is bf16 [4096,4096]
// with u = cols [0,DI) and z = cols [DI,2*DI). Output Ug fp32 [4096,DI].
// ---------------------------------------------------------------------------
__global__ void conv_silu_kernel(const bf16* __restrict__ XZ,
                                 const float* __restrict__ cw,
                                 const float* __restrict__ cb,
                                 float* __restrict__ Ug) {
    int idx = blockIdx.x * blockDim.x + threadIdx.x;      // row*DI + ch
    int ch  = idx & (DI_ - 1);
    int row = idx >> 11;                                  // DI=2048
    int b   = row >> 11;                                  // L=2048
    int l   = row & (L_ - 1);

    float acc = cb[ch];
#pragma unroll
    for (int j = 0; j < DC_; ++j) {
        int ls = l - (DC_ - 1) + j;
        if (ls >= 0)
            acc += (float)XZ[((size_t)(b * L_ + ls)) * NC_ + ch] * cw[ch * DC_ + j];
    }
    float z   = (float)XZ[(size_t)row * NC_ + DI_ + ch];
    float sil = z / (1.0f + __expf(-z));
    Ug[(size_t)row * DI_ + ch] = acc * sil;
}

// ---------------------------------------------------------------------------
// proj = Ug @ x_proj_w^T : [4096, 33]. One wave32 per output element.
// ---------------------------------------------------------------------------
__global__ void proj_kernel(const float* __restrict__ Ug,
                            const float* __restrict__ xw,
                            float* __restrict__ proj) {
    int gwave = (blockIdx.x * blockDim.x + threadIdx.x) >> 5;
    int lane  = threadIdx.x & 31;
    if (gwave >= ROWS_ * NPROJ_) return;
    int row = gwave / NPROJ_;
    int c   = gwave % NPROJ_;
    const float* u = Ug + (size_t)row * DI_;
    const float* w = xw + (size_t)c * DI_;
    float acc = 0.f;
    for (int k = lane; k < DI_; k += 32) acc += u[k] * w[k];
#pragma unroll
    for (int m = 16; m >= 1; m >>= 1) acc += __shfl_xor(acc, m, 32);
    if (lane == 0) proj[(size_t)row * NPROJ_ + c] = acc;
}

// ---------------------------------------------------------------------------
// Selective scan. One thread per (b, channel). h[16] in registers;
// delta/B/C rows staged through LDS in 32-timestep chunks.
// Writes y (incl. D*u skip) as bf16 for the output GEMM.
// ---------------------------------------------------------------------------
__global__ void scan_kernel(const float* __restrict__ Ug,
                            const float* __restrict__ proj,
                            const float* __restrict__ dtw,
                            const float* __restrict__ dtb,
                            const float* __restrict__ A_log,
                            const float* __restrict__ Dvec,
                            bf16* __restrict__ Yb) {
    const int b = blockIdx.x >> 3;                       // 8 blocks per batch
    const int i = (blockIdx.x & 7) * 256 + threadIdx.x;  // channel 0..2047

    float negA[DS_], h[DS_];
#pragma unroll
    for (int s = 0; s < DS_; ++s) {
        negA[s] = -__expf(A_log[(size_t)i * DS_ + s]);
        h[s] = 0.f;
    }
    const float wdt = dtw[i], bdt = dtb[i], Di = Dvec[i];

    __shared__ float pbuf[32 * NPROJ_];

    for (int t0 = 0; t0 < L_; t0 += 32) {
        __syncthreads();
        for (int k = threadIdx.x; k < 32 * NPROJ_; k += 256)
            pbuf[k] = proj[((size_t)(b * L_ + t0)) * NPROJ_ + k];
        __syncthreads();

        for (int tt = 0; tt < 32; ++tt) {
            const int t = t0 + tt;
            const float* pr = pbuf + tt * NPROJ_;
            const float u   = Ug[((size_t)(b * L_ + t)) * DI_ + i];
            const float din = pr[0] * wdt + bdt;
            const float sp  = (din > 20.f) ? din : log1pf(__expf(din)); // softplus
            const float dte = __expf(sp);                               // exp(softplus)
            float y = 0.f;
#pragma unroll
            for (int s = 0; s < DS_; ++s) {
                const float ab = __expf(negA[s] * dte);
                h[s] = ab * h[s] + pr[1 + s] * dte * u;
                y += h[s] * pr[1 + DS_ + s];
            }
            y += Di * u;
            Yb[((size_t)(b * L_ + t)) * DI_ + i] = (bf16)y;
        }
    }
}

// ---------------------------------------------------------------------------
// Launcher
// ---------------------------------------------------------------------------
extern "C" void kernel_launch(void* const* d_in, const int* in_sizes, int n_in,
                              void* d_out, int out_size, void* d_ws, size_t ws_size,
                              hipStream_t stream) {
    (void)in_sizes; (void)n_in; (void)out_size; (void)ws_size;

    const float* x          = (const float*)d_in[0];  // (B,L,DM)
    const float* in_proj_w  = (const float*)d_in[1];  // (2*DI, DM)
    const float* conv_w     = (const float*)d_in[2];  // (DI,1,DC)
    const float* conv_b     = (const float*)d_in[3];  // (DI,)
    const float* x_proj_w   = (const float*)d_in[4];  // (33, DI)
    const float* dt_proj_w  = (const float*)d_in[5];  // (DI,1)
    const float* dt_proj_b  = (const float*)d_in[6];  // (DI,)
    const float* A_log      = (const float*)d_in[7];  // (DI,DS)
    const float* Dv         = (const float*)d_in[8];  // (DI,)
    const float* out_proj_w = (const float*)d_in[9];  // (DM, DI)
    float* out = (float*)d_out;

    // workspace layout (bytes)
    char* ws = (char*)d_ws;
    const size_t MB = (size_t)1 << 20;
    bf16*  Xb   = (bf16*)(ws + 0 * MB);     // 4096*1024 bf16  (8 MB)
    bf16*  W1b  = (bf16*)(ws + 8 * MB);     // 4096*1024 bf16  (8 MB)
    bf16*  Wob  = (bf16*)(ws + 16 * MB);    // 1024*2048 bf16  (4 MB)
    bf16*  XZ   = (bf16*)(ws + 20 * MB);    // 4096*4096 bf16  (32 MB)
    float* Ug   = (float*)(ws + 52 * MB);   // 4096*2048 f32   (32 MB)
    float* proj = (float*)(ws + 84 * MB);   // 4096*33 f32     (<1 MB)
    bf16*  Yb   = (bf16*)(ws + 85 * MB);    // 4096*2048 bf16  (16 MB)

    // 1) fp32 -> bf16 conversions
    {
        int n = ROWS_ * DM_;  // 4,194,304 (x and in_proj_w same count)
        f32_to_bf16_kernel<<<(n + 255) / 256, 256, 0, stream>>>(x, Xb, n);
        f32_to_bf16_kernel<<<(n + 255) / 256, 256, 0, stream>>>(in_proj_w, W1b, n);
        int m = DM_ * DI_;    // 2,097,152
        f32_to_bf16_kernel<<<(m + 255) / 256, 256, 0, stream>>>(out_proj_w, Wob, m);
    }

    // 2) xz = x @ in_proj_w^T  : M=4096, N=4096, K=1024 -> bf16 XZ
    wmma_gemm_nt<true><<<dim3(NC_ / 128, ROWS_ / 64), 128, 0, stream>>>(
        Xb, W1b, nullptr, XZ, ROWS_, NC_, DM_);

    // 3) depthwise causal conv + silu gating -> Ug fp32
    {
        int n = ROWS_ * DI_;  // 8,388,608
        conv_silu_kernel<<<n / 256, 256, 0, stream>>>(XZ, conv_w, conv_b, Ug);
    }

    // 4) proj = Ug @ x_proj_w^T : [4096, 33]
    {
        int nwaves  = ROWS_ * NPROJ_;           // 135,168
        int nblocks = (nwaves + 7) / 8;         // 8 waves per 256-thread block
        proj_kernel<<<nblocks, 256, 0, stream>>>(Ug, x_proj_w, proj);
    }

    // 5) selective scan -> Yb bf16
    scan_kernel<<<16, 256, 0, stream>>>(Ug, proj, dt_proj_w, dt_proj_b, A_log, Dv, Yb);

    // 6) out = Y @ out_proj_w^T : M=4096, N=1024, K=2048 -> fp32 d_out
    wmma_gemm_nt<false><<<dim3(DM_ / 128, ROWS_ / 64), 128, 0, stream>>>(
        Yb, Wob, out, nullptr, ROWS_, DM_, DI_);
}
